// Down4D_70145405878301
// MI455X (gfx1250) — matrix operand order repaired and checked by reading, not compiled, not run on hardware
//
#include <hip/hip_runtime.h>
#include <hip/hip_bf16.h>

typedef __attribute__((ext_vector_type(16))) _Float16 v16h;
typedef __attribute__((ext_vector_type(8)))  float    v8f;

// ---------------- geometry ----------------
// x: (2,32,8,32,64,64) f32 ; pooled h: (2,32,4,16,32,32)
// padded f16 buffers (channels-fastest): [b][l+1][d+1][h+1][w+1][c]
//   hp : LP=6, DP=18, HP=34, WP=34, C=32  -> 7,990,272 halfs (15,980,544 B)
//   h2 : same dims, C=64                  -> 15,980,544 halfs (31,961,088 B)
// packed weights (WMMA A fragments, f16):
//   wp1: 4 co-tiles * 81 taps * 512        = 165,888 halfs (331,776 B)
//   wp2: 4 co-tiles * 81 taps * 2 kc * 512 = 331,776 halfs (663,552 B)

__global__ void zero_kernel(uint4* __restrict__ p, size_t n16) {
  size_t i = (size_t)blockIdx.x * blockDim.x + threadIdx.x;
  if (i < n16) p[i] = uint4{0u, 0u, 0u, 0u};
}

// Pack w1 (3,64,32,3,3,3) into WMMA A-fragment order.
// Fragment frag = co_tile*81 + tap; within fragment: lane*16 + e.
// A 16x32 f16 order used here (B matches): M = lane%16 ; K = e + 16*(lane>=16)
__global__ void pack_w1_kernel(const float* __restrict__ w, _Float16* __restrict__ wp) {
  int idx = blockIdx.x * 256 + threadIdx.x;          // 324*512 = 165,888 total
  if (idx >= 324 * 512) return;
  int e    = idx & 15;
  int lane = (idx >> 4) & 31;
  int frag = idx >> 9;
  int cot = frag / 81, tap = frag % 81;
  int kl = tap / 27, kd = (tap / 9) % 3, kh = (tap / 3) % 3, kw = tap % 3;
  int co = cot * 16 + (lane & 15);
  int ci = e + ((lane >> 4) * 16);
  size_t widx = ((((size_t)kl * 64 + co) * 32 + ci) * 3 + kd) * 9 + (size_t)kh * 3 + kw;
  wp[idx] = (_Float16)w[widx];
}

// Pack w2 (3,64,64,3,3,3); frag = (co_tile*81 + tap)*2 + kc ; ci = kc*32 + K
__global__ void pack_w2_kernel(const float* __restrict__ w, _Float16* __restrict__ wp) {
  int idx = blockIdx.x * 256 + threadIdx.x;          // 648*512 = 331,776 total
  if (idx >= 648 * 512) return;
  int e    = idx & 15;
  int lane = (idx >> 4) & 31;
  int frag = idx >> 9;
  int kc  = frag & 1;
  int ft  = frag >> 1;                                // co_tile*81 + tap
  int cot = ft / 81, tap = ft % 81;
  int kl = tap / 27, kd = (tap / 9) % 3, kh = (tap / 3) % 3, kw = tap % 3;
  int co = cot * 16 + (lane & 15);
  int ci = kc * 32 + e + ((lane >> 4) * 16);
  size_t widx = ((((size_t)kl * 64 + co) * 64 + ci) * 3 + kd) * 9 + (size_t)kh * 3 + kw;
  wp[idx] = (_Float16)w[widx];
}

// maxpool (L stride-2 slice + 2x2x2 pool) + f32->f16 + transpose to padded channels-fastest
__global__ void pool_kernel(const float* __restrict__ x, _Float16* __restrict__ hp) {
  int fl = blockIdx.x;                                // 4096 blocks: (b,l,d,h)
  int hh = fl & 31, dd = (fl >> 5) & 15, l = (fl >> 9) & 3, b = (fl >> 11) & 1;
  int w = threadIdx.x;                                // 0..31
  for (int c = threadIdx.y; c < 32; c += 8) {
    size_t base = (size_t)b * 33554432 + (size_t)c * 1048576 +
                  (size_t)(2 * l) * 131072 + (size_t)(2 * dd) * 4096 +
                  (size_t)(2 * hh) * 64 + (size_t)(2 * w);
    float m = -3.402823466e38f;
#pragma unroll
    for (int i = 0; i < 2; ++i)
#pragma unroll
      for (int j = 0; j < 2; ++j) {
        const float2 v2 = *(const float2*)(x + base + (size_t)i * 4096 + (size_t)j * 64);
        float v = v2.x > v2.y ? v2.x : v2.y;
        m = v > m ? v : m;
      }
    size_t off = (((((size_t)b * 6 + (l + 1)) * 18 + (dd + 1)) * 34 + (hh + 1)) * 34 + (w + 1)) * 32 + c;
    hp[off] = (_Float16)m;
  }
}

// conv1: Cin=32 -> Cout=64, 3x3x3x3 taps, bias-mask + ReLU, f16 out (padded layout)
// One block = full W row (N=32 -> 2 n-tiles); 2 waves, each owns 2 co-tiles x 2 n-tiles.
// Per tap: 2 A-loads + 2 B-loads -> 4 WMMAs (1 x 32B load per WMMA).
__global__ __launch_bounds__(64) void conv1_kernel(const _Float16* __restrict__ hp,
                                                   const _Float16* __restrict__ wp,
                                                   const float* __restrict__ bias,
                                                   _Float16* __restrict__ h2) {
  __shared__ __align__(128) _Float16 tile[27 * 34 * 32];          // 58,752 B
  int fl = blockIdx.x;                                // 4096 blocks: (b,l,d,h)
  int hh = fl & 31, dd = (fl >> 5) & 15, l = (fl >> 9) & 3, b = (fl >> 11) & 1;
  int tid = threadIdx.x;

  // stage input patch: 27 (kl,kd,kh) x 34 w-positions x 32 ci (64 B each)
  for (int p = tid; p < 918; p += 64) {
    int wi = p % 34; int q = p / 34;
    int kh = q % 3; q /= 3;
    int kd = q % 3; int kl = q / 3;
    size_t off = (((((size_t)b * 6 + (l + kl)) * 18 + (dd + kd)) * 34 + (hh + kh)) * 34 + wi) * 32;
    const uint4* src = (const uint4*)(hp + off);
    uint4* dst = (uint4*)(tile + p * 32);
    dst[0] = src[0]; dst[1] = src[1]; dst[2] = src[2]; dst[3] = src[3];
  }
  __syncthreads();

  int lane = tid & 31, wave = tid >> 5;               // wave = co-tile pair
  int n = lane & 15, hiHalf = lane >> 4;
  const _Float16* wbase0 = wp + ((size_t)wave * 2    ) * 81 * 512 + (size_t)lane * 16;
  const _Float16* wbase1 = wp + ((size_t)wave * 2 + 1) * 81 * 512 + (size_t)lane * 16;

  v8f acc00 = {}, acc01 = {}, acc10 = {}, acc11 = {};
  for (int tap = 0; tap < 81; ++tap) {
    int kl = tap / 27, kd = (tap / 9) % 3, kh = (tap / 3) % 3, kw = tap % 3;
    if (tap + 1 < 81) {
      __builtin_prefetch((const void*)(wbase0 + (tap + 1) * 512), 0, 3);
      __builtin_prefetch((const void*)(wbase1 + (tap + 1) * 512), 0, 3);
    }
    v16h a0 = *(const v16h*)(wbase0 + tap * 512);
    v16h a1 = *(const v16h*)(wbase1 + tap * 512);
    const _Float16* trow = tile + ((kl * 3 + kd) * 3 + kh) * 34 * 32 + hiHalf * 16;
    v16h b0 = *(const v16h*)(trow + (n + kw) * 32);
    v16h b1 = *(const v16h*)(trow + (n + 16 + kw) * 32);
    acc00 = __builtin_amdgcn_wmma_f32_16x16x32_f16(false, a0, false, b0, (short)0, acc00, false, false);
    acc01 = __builtin_amdgcn_wmma_f32_16x16x32_f16(false, a0, false, b1, (short)0, acc01, false, false);
    acc10 = __builtin_amdgcn_wmma_f32_16x16x32_f16(false, a1, false, b0, (short)0, acc10, false, false);
    acc11 = __builtin_amdgcn_wmma_f32_16x16x32_f16(false, a1, false, b1, (short)0, acc11, false, false);
  }

  // branch-free masked bias + ReLU + packed f16 stores (16 B per lane per tile)
  float m0 = (l > 0) ? 1.f : 0.f, m2 = (l < 3) ? 1.f : 0.f;
  int cobase0 = wave * 32 + hiHalf * 8;               // co-tile wave*2
  int cobase1 = cobase0 + 16;                         // co-tile wave*2+1
  float bs0[8], bs1[8];
#pragma unroll
  for (int r = 0; r < 8; ++r) {
    int c0 = cobase0 + r, c1 = cobase1 + r;
    bs0[r] = bias[64 + c0] + m0 * bias[c0] + m2 * bias[128 + c0];
    bs1[r] = bias[64 + c1] + m0 * bias[c1] + m2 * bias[128 + c1];
  }
  union { uint4 u; _Float16 h[8]; } p00, p01, p10, p11;
#pragma unroll
  for (int r = 0; r < 8; ++r) {
    float v;
    v = acc00[r] + bs0[r]; p00.h[r] = (_Float16)(v > 0.f ? v : 0.f);
    v = acc01[r] + bs0[r]; p01.h[r] = (_Float16)(v > 0.f ? v : 0.f);
    v = acc10[r] + bs1[r]; p10.h[r] = (_Float16)(v > 0.f ? v : 0.f);
    v = acc11[r] + bs1[r]; p11.h[r] = (_Float16)(v > 0.f ? v : 0.f);
  }
  size_t dbase = ((((size_t)b * 6 + (l + 1)) * 18 + (dd + 1)) * 34 + (hh + 1)) * 34;
  *(uint4*)(h2 + (dbase + n + 1)  * 64 + cobase0) = p00.u;
  *(uint4*)(h2 + (dbase + n + 17) * 64 + cobase0) = p01.u;
  *(uint4*)(h2 + (dbase + n + 1)  * 64 + cobase1) = p10.u;
  *(uint4*)(h2 + (dbase + n + 17) * 64 + cobase1) = p11.u;
}

// conv2: Cin=64 -> Cout=64; two K-phases re-using the same 58 KB LDS tile.
__global__ __launch_bounds__(64) void conv2_kernel(const _Float16* __restrict__ hs,
                                                   const _Float16* __restrict__ wp,
                                                   const float* __restrict__ bias,
                                                   float* __restrict__ out) {
  __shared__ __align__(128) _Float16 tile[27 * 34 * 32];          // 58,752 B
  int fl = blockIdx.x;                                // 4096 blocks: (b,l,d,h)
  int hh = fl & 31, dd = (fl >> 5) & 15, l = (fl >> 9) & 3, b = (fl >> 11) & 1;
  int tid = threadIdx.x;
  int lane = tid & 31, wave = tid >> 5;
  int n = lane & 15, hiHalf = lane >> 4;
  const _Float16* wbase0 = wp + ((size_t)wave * 2    ) * 81 * 2 * 512 + (size_t)lane * 16;
  const _Float16* wbase1 = wp + ((size_t)wave * 2 + 1) * 81 * 2 * 512 + (size_t)lane * 16;

  v8f acc00 = {}, acc01 = {}, acc10 = {}, acc11 = {};
  for (int kc = 0; kc < 2; ++kc) {
    // stage ci range [kc*32, kc*32+32) for all 918 positions
    for (int p = tid; p < 918; p += 64) {
      int wi = p % 34; int q = p / 34;
      int kh = q % 3; q /= 3;
      int kd = q % 3; int kl = q / 3;
      size_t off = (((((size_t)b * 6 + (l + kl)) * 18 + (dd + kd)) * 34 + (hh + kh)) * 34 + wi) * 64
                   + (size_t)kc * 32;
      const uint4* src = (const uint4*)(hs + off);
      uint4* dst = (uint4*)(tile + p * 32);
      dst[0] = src[0]; dst[1] = src[1]; dst[2] = src[2]; dst[3] = src[3];
    }
    __syncthreads();

    for (int tap = 0; tap < 81; ++tap) {
      int kl = tap / 27, kd = (tap / 9) % 3, kh = (tap / 3) % 3, kw = tap % 3;
      if (tap + 1 < 81) {
        __builtin_prefetch((const void*)(wbase0 + ((tap + 1) * 2 + kc) * 512), 0, 3);
        __builtin_prefetch((const void*)(wbase1 + ((tap + 1) * 2 + kc) * 512), 0, 3);
      }
      v16h a0 = *(const v16h*)(wbase0 + (tap * 2 + kc) * 512);
      v16h a1 = *(const v16h*)(wbase1 + (tap * 2 + kc) * 512);
      const _Float16* trow = tile + ((kl * 3 + kd) * 3 + kh) * 34 * 32 + hiHalf * 16;
      v16h b0 = *(const v16h*)(trow + (n + kw) * 32);
      v16h b1 = *(const v16h*)(trow + (n + 16 + kw) * 32);
      acc00 = __builtin_amdgcn_wmma_f32_16x16x32_f16(false, a0, false, b0, (short)0, acc00, false, false);
      acc01 = __builtin_amdgcn_wmma_f32_16x16x32_f16(false, a0, false, b1, (short)0, acc01, false, false);
      acc10 = __builtin_amdgcn_wmma_f32_16x16x32_f16(false, a1, false, b0, (short)0, acc10, false, false);
      acc11 = __builtin_amdgcn_wmma_f32_16x16x32_f16(false, a1, false, b1, (short)0, acc11, false, false);
    }
    __syncthreads();   // tile re-staged next phase
  }

  float m0 = (l > 0) ? 1.f : 0.f, m2 = (l < 3) ? 1.f : 0.f;
  int cobase0 = wave * 32 + hiHalf * 8;
  int cobase1 = cobase0 + 16;
  size_t obase = (size_t)b * 4194304 + (size_t)l * 16384 + (size_t)dd * 1024 + (size_t)hh * 32;
#pragma unroll
  for (int r = 0; r < 8; ++r) {
    int c0 = cobase0 + r, c1 = cobase1 + r;
    float bsA = bias[64 + c0] + m0 * bias[c0] + m2 * bias[128 + c0];
    float bsB = bias[64 + c1] + m0 * bias[c1] + m2 * bias[128 + c1];
    float v;
    v = acc00[r] + bsA; out[obase + (size_t)c0 * 65536 + n]      = v > 0.f ? v : 0.f;
    v = acc01[r] + bsA; out[obase + (size_t)c0 * 65536 + n + 16] = v > 0.f ? v : 0.f;
    v = acc10[r] + bsB; out[obase + (size_t)c1 * 65536 + n]      = v > 0.f ? v : 0.f;
    v = acc11[r] + bsB; out[obase + (size_t)c1 * 65536 + n + 16] = v > 0.f ? v : 0.f;
  }
}

extern "C" void kernel_launch(void* const* d_in, const int* in_sizes, int n_in,
                              void* d_out, int out_size, void* d_ws, size_t ws_size,
                              hipStream_t stream) {
  (void)in_sizes; (void)n_in; (void)out_size; (void)ws_size;
  const float* x  = (const float*)d_in[0];
  const float* w1 = (const float*)d_in[1];
  const float* b1 = (const float*)d_in[2];
  const float* w2 = (const float*)d_in[3];
  const float* b2 = (const float*)d_in[4];
  float* out = (float*)d_out;

  char* ws = (char*)d_ws;
  _Float16* hp  = (_Float16*)(ws);                    // 15,980,544 B
  _Float16* h2  = (_Float16*)(ws + 15980544);         // 31,961,088 B
  _Float16* wp1 = (_Float16*)(ws + 47941632);         //    331,776 B
  _Float16* wp2 = (_Float16*)(ws + 48273408);         //    663,552 B

  // zero padded activation buffers (halos must be 0 every call)
  zero_kernel<<<11705, 256, 0, stream>>>((uint4*)ws, 2996352);   // 47,941,632 B
  pack_w1_kernel<<<648, 256, 0, stream>>>(w1, wp1);
  pack_w2_kernel<<<1296, 256, 0, stream>>>(w2, wp2);
  pool_kernel<<<4096, dim3(32, 8), 0, stream>>>(x, hp);
  conv1_kernel<<<4096, 64, 0, stream>>>(hp, wp1, b1, h2);
  conv2_kernel<<<4096, 64, 0, stream>>>(h2, wp2, b2, out);
}
